// GistSlotAttention_50646254355055
// MI455X (gfx1250) — matrix operand order repaired and checked by reading, not compiled, not run on hardware
//
#include <hip/hip_runtime.h>
#include <hip/hip_bf16.h>
#include <math.h>

// ---------------------------------------------------------------------------
// CDNA5 (gfx1250) WMMA types & helpers (wave32)
// ---------------------------------------------------------------------------
typedef __bf16 bf16;
typedef __attribute__((ext_vector_type(16))) __bf16 bf16x16;
typedef __attribute__((ext_vector_type(8)))  float  f32x8;

#define DEV static __device__ __forceinline__

DEV int lane_id() { return (int)(threadIdx.x & 31); }

// A fragment: 16(M) x 32(K) bf16, element (i,k) at p[i*sRow + k*sK]
// Layout (ISA 7.12.2, 16-bit A 16x32): lanes 0-15 -> K {0..7,16..23},
// lanes 16-31 -> K {8..15,24..31}; vgpr j packs 2 consecutive K.
template <typename T>
DEV bf16x16 frag_a(const T* p, int sRow, int sK, int k0) {
  int l = lane_id();
  int r = l & 15;
  int hi = l >> 4;
  const T* q0 = p + (size_t)r * sRow + (size_t)(k0 + hi * 8) * sK;
  const T* q1 = q0 + (size_t)16 * sK;
  bf16x16 f;
#pragma unroll
  for (int j = 0; j < 8; ++j) {
    f[j]     = (bf16)(float)q0[(size_t)j * sK];
    f[j + 8] = (bf16)(float)q1[(size_t)j * sK];
  }
  return f;
}

// B fragment: 32(K) x 16(N) bf16, element (k,c) at p[c*sCol + k*sK]
// Layout (same K order as A): lanes 0-15 hold K {0..7,16..23} of col=lane,
// lanes 16-31 hold K {8..15,24..31} of col=lane-16.
template <typename T>
DEV bf16x16 frag_b(const T* p, int sCol, int sK, int k0) {
  int l = lane_id();
  int c = l & 15;
  int hi = l >> 4;
  const T* q0 = p + (size_t)c * sCol + (size_t)(k0 + hi * 8) * sK;
  const T* q1 = q0 + (size_t)16 * sK;
  bf16x16 f;
#pragma unroll
  for (int j = 0; j < 8; ++j) {
    f[j]     = (bf16)(float)q0[(size_t)j * sK];
    f[j + 8] = (bf16)(float)q1[(size_t)j * sK];
  }
  return f;
}

DEV f32x8 wmma_bf16(bf16x16 a, bf16x16 b, f32x8 c) {
  return __builtin_amdgcn_wmma_f32_16x16x32_bf16(
      /*neg_a=*/false, a, /*neg_b=*/false, b,
      /*c_mod=*/(short)0, c, /*reuse_a=*/false, /*reuse_b=*/false);
}

// C/D 16x16 f32: vgpr v -> M = v + 8*(lane>=16), N = lane&15
template <typename T>
DEV void store_c_t(T* p, int ldc, f32x8 acc) {
  int l = lane_id();
  int n = l & 15;
  int hi = l >> 4;
#pragma unroll
  for (int v = 0; v < 8; ++v) p[(size_t)(v + 8 * hi) * ldc + n] = (T)acc[v];
}

DEV f32x8 load_c(const float* p, int ldc) {
  int l = lane_id();
  int n = l & 15;
  int hi = l >> 4;
  f32x8 c;
#pragma unroll
  for (int v = 0; v < 8; ++v) c[v] = p[(size_t)(v + 8 * hi) * ldc + n];
  return c;
}

DEV float silu_f(float x) { return x / (1.f + __expf(-x)); }

// ---------------------------------------------------------------------------
// CDNA5 async Memory -> LDS copies (ASYNCcnt tracked), inline asm: no builtin
// exists for GLOBAL_LOAD_ASYNC_TO_LDS_* (ISA 15.18.3 opcodes 96..98).
// lds_off is a wave-relative LDS byte offset.
// ---------------------------------------------------------------------------
DEV void async_load_b128(unsigned lds_off, unsigned long long gaddr) {
  asm volatile("global_load_async_to_lds_b128 %0, %1, off"
               :: "v"(lds_off), "v"(gaddr)
               : "memory");
}
DEV void async_load_b64(unsigned lds_off, unsigned long long gaddr) {
  asm volatile("global_load_async_to_lds_b64 %0, %1, off"
               :: "v"(lds_off), "v"(gaddr)
               : "memory");
}
DEV void wait_async0() { asm volatile("s_wait_asynccnt 0" ::: "memory"); }

// ---------------------------------------------------------------------------
// Kernel 1: Y[M,N] = act(X[M,K] @ W[N,K]^T)   (bf16 WMMA, f32 accumulate)
// block 256 threads = 8 waves; block tile 64(M) x 128(N); wave tile 32x32.
// OutT = bf16 for q/k/v projections (halves HBM traffic; all consumers are
// bf16 WMMA anyway), float for the final output projection.
// ---------------------------------------------------------------------------
template <typename OutT, int ACT>
__global__ __launch_bounds__(256) void gemm_xwT(const float* __restrict__ X,
                                                const float* __restrict__ W,
                                                OutT* __restrict__ Y, int K,
                                                int N) {
  int n0 = blockIdx.x * 128;
  int m0 = blockIdx.y * 64;
  int w = threadIdx.x >> 5;
  int mw = m0 + (w & 1) * 32;
  int nw = n0 + (w >> 1) * 32;

  f32x8 acc[2][2] = {};
  for (int k0 = 0; k0 < K; k0 += 32) {
    bf16x16 a0 = frag_a(X + (size_t)mw * K, K, 1, k0);
    bf16x16 a1 = frag_a(X + (size_t)(mw + 16) * K, K, 1, k0);
    bf16x16 b0 = frag_b(W + (size_t)nw * K, K, 1, k0);
    bf16x16 b1 = frag_b(W + (size_t)(nw + 16) * K, K, 1, k0);
    acc[0][0] = wmma_bf16(a0, b0, acc[0][0]);
    acc[0][1] = wmma_bf16(a0, b1, acc[0][1]);
    acc[1][0] = wmma_bf16(a1, b0, acc[1][0]);
    acc[1][1] = wmma_bf16(a1, b1, acc[1][1]);
  }
#pragma unroll
  for (int i = 0; i < 2; ++i)
#pragma unroll
    for (int j = 0; j < 2; ++j) {
      f32x8 r = acc[i][j];
      if (ACT) {
#pragma unroll
        for (int v = 0; v < 8; ++v) r[v] = silu_f(r[v]);
      }
      store_c_t(Y + (size_t)(mw + i * 16) * N + (nw + j * 16), N, r);
    }
}

// ---------------------------------------------------------------------------
// Kernel 2: gate  s[b,t,h,m] = sigmoid( dot(q[b,gs+m,h,:], k[b,t,h,:]) ) / 8
// grid (T/64, B*H); block tile 64(t) x 64(m); wave tile 16x32. q/k are bf16.
// ---------------------------------------------------------------------------
__global__ __launch_bounds__(256) void gate_kernel(const bf16* __restrict__ q,
                                                   const bf16* __restrict__ k,
                                                   const int* __restrict__ gsp,
                                                   float* __restrict__ s,
                                                   int T) {
  int t0 = blockIdx.x * 64;
  int bh = blockIdx.y;
  int b = bh >> 2;  // H = 4
  int h = bh & 3;
  int gs = *gsp;

  const bf16* A = k + ((size_t)b * T + t0) * 1024 + h * 256;   // rows t
  const bf16* Bq = q + ((size_t)b * T + gs) * 1024 + h * 256;  // rows m

  int w = threadIdx.x >> 5;
  int tl = (w & 3) * 16;   // local t tile
  int ml = (w >> 2) * 32;  // local m tile

  f32x8 acc[2] = {};
  for (int k0 = 0; k0 < 256; k0 += 32) {
    bf16x16 a = frag_a(A + (size_t)tl * 1024, 1024, 1, k0);
    bf16x16 b0 = frag_b(Bq + (size_t)ml * 1024, 1024, 1, k0);
    bf16x16 b1 = frag_b(Bq + (size_t)(ml + 16) * 1024, 1024, 1, k0);
    acc[0] = wmma_bf16(a, b0, acc[0]);
    acc[1] = wmma_bf16(a, b1, acc[1]);
  }
#pragma unroll
  for (int j = 0; j < 2; ++j) {
    f32x8 r = acc[j];
#pragma unroll
    for (int v = 0; v < 8; ++v) r[v] = 0.125f / (1.f + __expf(-r[v]));
    // s layout [B,T,H,64]; t-row stride = H*64 = 256
    store_c_t(s + ((size_t)b * T + t0 + tl) * 256 + h * 64 + ml + j * 16, 256,
              r);
  }
}

// ---------------------------------------------------------------------------
// Kernel 3: chunked gated-slot-attention recurrence, one workgroup per (b,h).
// Chunk C=64; state HK[256,64], HV[64,256] f32 in LDS (CDNA5: 320KB/WGP).
// Chunk tiles arrive via GLOBAL_LOAD_ASYNC_TO_LDS (ASYNCcnt). ~296 KB LDS.
// ---------------------------------------------------------------------------
__global__ __launch_bounds__(256) void gsa_kernel(const bf16* __restrict__ q,
                                                  const bf16* __restrict__ k,
                                                  const bf16* __restrict__ v,
                                                  const float* __restrict__ s,
                                                  float* __restrict__ o,
                                                  int T) {
  constexpr int C = 64;
  int b = blockIdx.x >> 2;
  int h = blockIdx.x & 3;
  int tid = threadIdx.x;
  int w = tid >> 5;
  int lane = tid & 31;

  extern __shared__ char smem_raw[];
  float* HK = (float*)smem_raw;       // [256][64]
  float* HV = HK + 256 * 64;          // [64][256]
  float* Pf = HV + 64 * 256;          // [64][64] cumulative decay
  float* sOK = Pf + 64 * 64;          // [64][64] s chunk, then ok
  float* Mf = sOK + 64 * 64;          // [64][64] scratch (S)
  bf16* Qb = (bf16*)(Mf + 64 * 64);   // [64][256]
  bf16* Kb = Qb + 64 * 256;           // [64][256]
  bf16* Vb = Kb + 64 * 256;           // [64][256]
  bf16* Ub = Vb + 64 * 256;           // [64][64]  U = s/P
  bf16* Sb = Ub + 64 * 64;            // [64][64]  tril(S), then tril(W U^T)
  bf16* Wb = Sb + 64 * 64;            // [64][64]  W = qv * P

  for (int i = tid; i < 256 * 64; i += 256) {
    HK[i] = 0.f;
    HV[i] = 0.f;
  }
  __syncthreads();

  for (int t0 = 0; t0 < T; t0 += C) {
    const bf16* qg = q + ((size_t)b * T + t0) * 1024 + h * 256;
    const bf16* kg = k + ((size_t)b * T + t0) * 1024 + h * 256;
    const bf16* vg = v + ((size_t)b * T + t0) * 1024 + h * 256;
    const float* sg = s + ((size_t)b * T + t0) * 256 + h * 64;

    // ---- async-copy chunk tiles into LDS (raw bf16 / f32 bytes) ----
    // Q/K/V: 3 tiles x 64 rows; each row = 512B = one B128 per lane.
    {
      unsigned lb = (unsigned)(lane * 16);
      for (int r = w; r < 192; r += 8) {
        int tile = r >> 6;
        int row = r & 63;
        const bf16* gsrc = (tile == 0 ? qg : tile == 1 ? kg : vg) +
                           (size_t)row * 1024;
        bf16* ldst = (tile == 0 ? Qb : tile == 1 ? Kb : Vb) + row * 256;
        async_load_b128((unsigned)((const char*)ldst - smem_raw) + lb,
                        (unsigned long long)(size_t)((const char*)gsrc) + lb);
      }
      // s chunk: 64 rows x 256B = one B64 per lane per row.
      unsigned lb8 = (unsigned)(lane * 8);
      for (int r = w; r < 64; r += 8) {
        const float* gsrc = sg + (size_t)r * 256;
        float* ldst = sOK + r * 64;
        async_load_b64((unsigned)((const char*)ldst - smem_raw) + lb8,
                       (unsigned long long)(size_t)((const char*)gsrc) + lb8);
      }
      wait_async0();
    }
    __syncthreads();

    // ---- per-slot cumulative decay P and U = s/P ----
    if (tid < 64) {
      int m = tid;
      float p = 1.f;
      for (int t = 0; t < C; ++t) {
        float sv = sOK[t * 64 + m];
        p *= (1.f - sv);
        Pf[t * 64 + m] = p;
        Ub[t * 64 + m] = (bf16)(sv / p);
      }
    }
    __syncthreads();

    // ---- S = Q @ K^T  -> Mf ----
    for (int jj = 0; jj < 2; ++jj) {
      int job = w + jj * 8;
      int rt = job >> 2, ct = job & 3;
      f32x8 acc = {};
      for (int k0 = 0; k0 < 256; k0 += 32) {
        bf16x16 a = frag_a(Qb + rt * 16 * 256, 256, 1, k0);
        bf16x16 bb = frag_b(Kb + ct * 16 * 256, 256, 1, k0);  // (kk,tau)
        acc = wmma_bf16(a, bb, acc);
      }
      store_c_t(Mf + rt * 16 * 64 + ct * 16, 64, acc);
    }
    __syncthreads();

    // ---- mask tril (tau <= t) -> Sb (bf16) ----
    for (int e = tid; e < 64 * 64; e += 256) {
      int t = e >> 6, ta = e & 63;
      Sb[e] = (bf16)((ta <= t) ? Mf[e] : 0.f);
    }
    __syncthreads();

    // ---- ok = P * (Q @ HK0 + tril(S) @ U) -> sOK ----
    for (int jj = 0; jj < 2; ++jj) {
      int job = w + jj * 8;
      int rt = job >> 2, ct = job & 3;
      f32x8 acc = {};
      for (int k0 = 0; k0 < 256; k0 += 32) {
        bf16x16 a = frag_a(Qb + rt * 16 * 256, 256, 1, k0);
        bf16x16 bb = frag_b(HK + ct * 16, 1, 64, k0);  // (kk,m)=HK[kk][m] f32
        acc = wmma_bf16(a, bb, acc);
      }
      for (int k0 = 0; k0 < 64; k0 += 32) {
        bf16x16 a = frag_a(Sb + rt * 16 * 64, 64, 1, k0);
        bf16x16 bb = frag_b(Ub + ct * 16, 1, 64, k0);  // (tau,m)=U[tau][m]
        acc = wmma_bf16(a, bb, acc);
      }
      int n = lane & 15, hi = lane >> 4;
#pragma unroll
      for (int vv = 0; vv < 8; ++vv) {
        int row = rt * 16 + vv + 8 * hi;
        int col = ct * 16 + n;
        acc[vv] *= Pf[row * 64 + col];
      }
      store_c_t(sOK + rt * 16 * 64 + ct * 16, 64, acc);
    }
    __syncthreads();

    // ---- softmax over slots, W = qv * P -> Wb ----
    if (tid < 64) {
      float* row = sOK + tid * 64;
      float mx = -1e30f;
      for (int m = 0; m < 64; ++m) mx = fmaxf(mx, row[m]);
      float sum = 0.f;
      for (int m = 0; m < 64; ++m) {
        float e = __expf(row[m] - mx);
        row[m] = e;
        sum += e;
      }
      float inv = 1.f / sum;
      for (int m = 0; m < 64; ++m)
        Wb[tid * 64 + m] = (bf16)(row[m] * inv * Pf[tid * 64 + m]);
    }
    __syncthreads();

    // ---- G = tril(W @ U^T) -> Sb ----
    for (int jj = 0; jj < 2; ++jj) {
      int job = w + jj * 8;
      int rt = job >> 2, ct = job & 3;
      f32x8 acc = {};
      for (int k0 = 0; k0 < 64; k0 += 32) {
        bf16x16 a = frag_a(Wb + rt * 16 * 64, 64, 1, k0);
        bf16x16 bb = frag_b(Ub + ct * 16 * 64, 64, 1, k0);  // (m,tau)=U[tau][m]
        acc = wmma_bf16(a, bb, acc);
      }
      int n = lane & 15, hi = lane >> 4;
#pragma unroll
      for (int vv = 0; vv < 8; ++vv) {
        int row = rt * 16 + vv + 8 * hi;  // t
        int col = ct * 16 + n;            // tau
        Sb[row * 64 + col] = (bf16)((col <= row) ? acc[vv] : 0.f);
      }
    }
    __syncthreads();

    // ---- O = W @ HV0 + G @ V  -> global ----
    for (int jj = 0; jj < 8; ++jj) {
      int job = w * 8 + jj;
      int rt = job >> 4, ct = job & 15;
      f32x8 acc = {};
      for (int k0 = 0; k0 < 64; k0 += 32) {
        bf16x16 a = frag_a(Wb + rt * 16 * 64, 64, 1, k0);
        bf16x16 bb = frag_b(HV + ct * 16, 1, 256, k0);  // (m,v)=HV[m][v] f32
        acc = wmma_bf16(a, bb, acc);
      }
      for (int k0 = 0; k0 < 64; k0 += 32) {
        bf16x16 a = frag_a(Sb + rt * 16 * 64, 64, 1, k0);
        bf16x16 bb = frag_b(Vb + ct * 16, 1, 256, k0);  // (tau,v)=V[tau][v]
        acc = wmma_bf16(a, bb, acc);
      }
      store_c_t(o + ((size_t)b * T + t0 + rt * 16) * 1024 + h * 256 + ct * 16,
                1024, acc);
    }
    __syncthreads();

    // ---- HK = P_C * (HK0 + K^T @ U) ----
    for (int jj = 0; jj < 8; ++jj) {
      int job = w * 8 + jj;
      int rt = job >> 2, ct = job & 3;  // rows kk (16x), cols m (4x)
      f32x8 acc = load_c(HK + rt * 16 * 64 + ct * 16, 64);
      for (int k0 = 0; k0 < 64; k0 += 32) {
        bf16x16 a = frag_a(Kb + rt * 16, 1, 256, k0);  // (kk,tau)=K[tau][kk]
        bf16x16 bb = frag_b(Ub + ct * 16, 1, 64, k0);  // (tau,m)
        acc = wmma_bf16(a, bb, acc);
      }
      float pc = Pf[63 * 64 + ct * 16 + (lane & 15)];
#pragma unroll
      for (int vv = 0; vv < 8; ++vv) acc[vv] *= pc;
      store_c_t(HK + rt * 16 * 64 + ct * 16, 64, acc);
    }
    // ---- HV = P_C * (HV0 + U^T @ V) ----
    for (int jj = 0; jj < 8; ++jj) {
      int job = w * 8 + jj;
      int rt = job >> 4, ct = job & 15;  // rows m (4x), cols v (16x)
      f32x8 acc = load_c(HV + rt * 16 * 256 + ct * 16, 256);
      for (int k0 = 0; k0 < 64; k0 += 32) {
        bf16x16 a = frag_a(Ub + rt * 16, 1, 64, k0);    // (m,tau)=U[tau][m]
        bf16x16 bb = frag_b(Vb + ct * 16, 1, 256, k0);  // (tau,v)
        acc = wmma_bf16(a, bb, acc);
      }
      int hi = lane >> 4;
#pragma unroll
      for (int vv = 0; vv < 8; ++vv)
        acc[vv] *= Pf[63 * 64 + rt * 16 + vv + 8 * hi];
      store_c_t(HV + rt * 16 * 256 + ct * 16, 256, acc);
    }
    __syncthreads();
  }
}

// ---------------------------------------------------------------------------
// Kernel 4: y = rmsnorm(silu(o)) * g_norm_weight   (one block per row of 1024)
// ---------------------------------------------------------------------------
__global__ __launch_bounds__(256) void norm_kernel(const float* __restrict__ o,
                                                   const float* __restrict__ gw,
                                                   float* __restrict__ y) {
  __shared__ float red[256];
  size_t row = blockIdx.x;
  const float* orow = o + row * 1024;
  float vals[4];
  float ss = 0.f;
#pragma unroll
  for (int j = 0; j < 4; ++j) {
    float x = orow[threadIdx.x + j * 256];
    x = silu_f(x);
    vals[j] = x;
    ss += x * x;
  }
  red[threadIdx.x] = ss;
  __syncthreads();
  for (int off = 128; off > 0; off >>= 1) {
    if ((int)threadIdx.x < off) red[threadIdx.x] += red[threadIdx.x + off];
    __syncthreads();
  }
  float scale = rsqrtf(red[0] * (1.f / 1024.f) + 1e-5f);
#pragma unroll
  for (int j = 0; j < 4; ++j) {
    int c = threadIdx.x + j * 256;
    y[row * 1024 + c] = vals[j] * scale * gw[c];
  }
}

// ---------------------------------------------------------------------------
// Launch
// ---------------------------------------------------------------------------
extern "C" void kernel_launch(void* const* d_in, const int* in_sizes, int n_in,
                              void* d_out, int out_size, void* d_ws,
                              size_t ws_size, hipStream_t stream) {
  const float* hs = (const float*)d_in[0];
  const float* Wq = (const float*)d_in[1];
  const float* Wk = (const float*)d_in[2];
  const float* Wv = (const float*)d_in[3];
  const float* gw = (const float*)d_in[4];
  const float* Wo = (const float*)d_in[5];
  const int* gist_start = (const int*)d_in[6];

  const int B = 4, T = 2048, D = 1024;
  const size_t MT = (size_t)B * T;  // 8192 rows

  // workspace carve: q/k/v bf16, s/o/y f32
  bf16* qb = (bf16*)d_ws;              // [MT,1024] bf16
  bf16* kb = qb + MT * D;              // [MT,1024] bf16
  bf16* vb = kb + MT * D;              // [MT,1024] bf16
  float* sb = (float*)(vb + MT * D);   // [B,T,H,64] = [MT,256] f32
  float* ob = sb + MT * 256;           // [MT,1024] f32
  float* yb = ob + MT * D;             // [MT,1024] f32

  dim3 blk(256);
  dim3 gA(D / 128, (unsigned)(MT / 64));

  gemm_xwT<bf16, 1><<<gA, blk, 0, stream>>>(hs, Wq, qb, D, D);
  gemm_xwT<bf16, 1><<<gA, blk, 0, stream>>>(hs, Wk, kb, D, D);
  gemm_xwT<bf16, 1><<<gA, blk, 0, stream>>>(hs, Wv, vb, D, D);

  gate_kernel<<<dim3(T / 64, B * 4), blk, 0, stream>>>(qb, kb, gist_start, sb,
                                                       T);

  // HK/HV f32 states + bf16 chunk tiles: 303104 bytes of LDS (<320KB/WGP)
  const size_t shmem = (size_t)(256 * 64 + 64 * 256 + 3 * 64 * 64) * 4 +
                       (size_t)(3 * 64 * 256 + 3 * 64 * 64) * 2;
  (void)hipFuncSetAttribute((const void*)gsa_kernel,
                            hipFuncAttributeMaxDynamicSharedMemorySize,
                            (int)shmem);
  gsa_kernel<<<dim3(B * 4), blk, shmem, stream>>>(qb, kb, vb, sb, ob, T);

  norm_kernel<<<dim3((unsigned)MT), blk, 0, stream>>>(ob, gw, yb);

  gemm_xwT<float, 0><<<gA, blk, 0, stream>>>(yb, Wo, (float*)d_out, D, D);
}
